// MiniSelectiveSSM_86225763434829
// MI455X (gfx1250) — matrix operand order repaired and checked by reading, not compiled, not run on hardware
//
#include <hip/hip_runtime.h>
#include <hip/hip_bf16.h>

// ---------------------------------------------------------------------------
// MiniSelectiveSSM for gfx1250 (MI455X): bf16 WMMA GEMMs + chunked linear scan
//   B=8, T=2048, D=1024, N=256  (M = B*T = 16384)
// ---------------------------------------------------------------------------

typedef __attribute__((ext_vector_type(16))) __bf16 bf16x16;
typedef __attribute__((ext_vector_type(8)))  __bf16 bf16x8;
typedef __attribute__((ext_vector_type(8)))  float  v8f;
typedef unsigned short u16;
typedef unsigned int   u32;

// float -> bf16 round-to-nearest-even
__device__ __forceinline__ u16 f2bf(float f) {
    u32 u = __float_as_uint(f);
    u += 0x7fffu + ((u >> 16) & 1u);
    return (u16)(u >> 16);
}

// Load a 16-element bf16 fragment as two 16-byte chunks `gap` elements apart.
__device__ __forceinline__ bf16x16 ld_frag(const u16* p, int gap) {
    bf16x8 lo = *(const bf16x8*)(p);
    bf16x8 hi = *(const bf16x8*)(p + gap);
    return __builtin_shufflevector(lo, hi, 0, 1, 2, 3, 4, 5, 6, 7,
                                           8, 9, 10, 11, 12, 13, 14, 15);
}

// ---------------------------------------------------------------------------
// 1) Elementwise fp32 -> bf16 conversion (for x)
// ---------------------------------------------------------------------------
__global__ __launch_bounds__(256) void convert_f32_bf16(
    const float* __restrict__ src, u16* __restrict__ dst, int n)
{
    int i = blockIdx.x * blockDim.x + threadIdx.x;
    int stride = gridDim.x * blockDim.x;
    for (; i < n; i += stride) dst[i] = f2bf(src[i]);
}

// ---------------------------------------------------------------------------
// 2) Transpose + convert weights: src is (rowsK x cols) fp32, dst is
//    (cols x rowsK) bf16, so GEMM B-fragments are contiguous along K.
// ---------------------------------------------------------------------------
__global__ __launch_bounds__(256) void transpose_to_bf16(
    const float* __restrict__ src, u16* __restrict__ dst, int rowsK, int cols)
{
    int i = blockIdx.x * blockDim.x + threadIdx.x;
    int total = rowsK * cols;
    int stride = gridDim.x * blockDim.x;
    for (; i < total; i += stride) {
        int k = i / cols;
        int n = i - k * cols;
        dst[(size_t)n * rowsK + k] = f2bf(src[i]);
    }
}

// ---------------------------------------------------------------------------
// 3) bf16 WMMA GEMM:  out(MxN) = act( A(MxK) * B(KxN) + bias ) [+ resid]
//    A row-major bf16; B supplied transposed as Bt(NxK) bf16.
//    Block: 256 threads = 8 waves arranged 4(rows) x 2(cols);
//    each wave owns a 32x64 tile = 8 accumulators of v_wmma_f32_16x16x32_bf16
//    (2 A-fragments x 4 B-fragments per K-step -> one load clause, 8 WMMAs).
//    Block tile = 128x128. Requires M%128==0, N%128==0, K%32==0 (all hold).
//    act: 0=none, 1=sigmoid, 2=tanh
// ---------------------------------------------------------------------------
__global__ __launch_bounds__(256) void gemm_bf16_wmma(
    const u16*  __restrict__ A,
    const u16*  __restrict__ Bt,
    const float* __restrict__ bias,
    const float* __restrict__ resid,   // may be nullptr (also may alias out)
    float* __restrict__ out,
    int M, int N, int K, int act)
{
    const int lane = threadIdx.x & 31;
    const int wave = threadIdx.x >> 5;
    const int half = lane >> 4;     // which 16-lane half of the wave
    const int r    = lane & 15;
    const int wr   = wave & 3;      // row group within block (0..3)
    const int wc   = wave >> 2;     // col group within block (0..1)
    const int row0 = blockIdx.x * 128 + wr * 32;
    const int col0 = blockIdx.y * 128 + wc * 64;

    v8f acc[8];                     // [c] rows row0.., [4+c] rows row0+16..
#pragma unroll
    for (int i = 0; i < 8; ++i) acc[i] = v8f{};

    // A fragment (16-bit A 16x32 layout, ISA 7.12.2): lane half h, row r holds
    //   K = k0 + 8h + [0..7]  and  K = k0 + 16 + 8h + [0..7]
    const u16* aRowL = A + (size_t)(row0 +      r) * K + half * 8;
    const u16* aRowH = A + (size_t)(row0 + 16 + r) * K + half * 8;
    // B fragment (16-bit B layout): lane half h, column n holds
    //   K = k0 + 8h + [0..7]  and  K = k0 + 16 + 8h + [0..7] (Bt row-major along K).
    const u16* bRows[4];
#pragma unroll
    for (int c = 0; c < 4; ++c)
        bRows[c] = Bt + (size_t)(col0 + c * 16 + r) * K + half * 8;

    for (int k0 = 0; k0 < K; k0 += 32) {
        // Speculative prefetch of the next K-step (dropped if OOB).
        __builtin_prefetch((const void*)(aRowL + k0 + 32), 0, 1);
        __builtin_prefetch((const void*)(aRowH + k0 + 32), 0, 1);
        __builtin_prefetch((const void*)(bRows[0] + k0 + 32), 0, 1);
        __builtin_prefetch((const void*)(bRows[2] + k0 + 32), 0, 1);

        // Load ALL fragments into distinct registers first so the scheduler
        // can clause the loads and issue the 8 WMMAs back-to-back.
        bf16x16 afL = ld_frag(aRowL + k0, 16);
        bf16x16 afH = ld_frag(aRowH + k0, 16);
        bf16x16 bfr[4];
#pragma unroll
        for (int c = 0; c < 4; ++c) bfr[c] = ld_frag(bRows[c] + k0, 16);

#pragma unroll
        for (int c = 0; c < 4; ++c)
            acc[c] = __builtin_amdgcn_wmma_f32_16x16x32_bf16(
                false, afL, false, bfr[c], (short)0, acc[c], false, false);
#pragma unroll
        for (int c = 0; c < 4; ++c)
            acc[4 + c] = __builtin_amdgcn_wmma_f32_16x16x32_bf16(
                false, afH, false, bfr[c], (short)0, acc[4 + c], false, false);
    }

    // Epilogue. D layout: vgpr i, lane half h -> M = row0 + 16g + 8h + i,
    // N = col0 + 16c + r.
#pragma unroll
    for (int g = 0; g < 2; ++g) {
        const int rowBase = row0 + g * 16 + half * 8;
#pragma unroll
        for (int c = 0; c < 4; ++c) {
            const int n  = col0 + c * 16 + r;
            const float bv = bias[n];
#pragma unroll
            for (int i = 0; i < 8; ++i) {
                const int m = rowBase + i;
                float v = acc[g * 4 + c][i] + bv;
                if (act == 1)      v = 1.0f / (1.0f + expf(-v));
                else if (act == 2) v = tanhf(v);
                if (resid) v += resid[(size_t)m * N + n];
                out[(size_t)m * N + n] = v;
            }
        }
    }
}

// ---------------------------------------------------------------------------
// 4) Chunked linear scan:  s_t = a_t*s_{t-1} + (1-a_t)*b_t,  emit c_t*s_t.
//    The recurrence is affine => associative. 3 passes over P=T/L chunks.
//    Layouts: a,b,c fp32 (B,T,N); cs bf16 (B,T,N). Channel ch = b*N + n.
// ---------------------------------------------------------------------------
__global__ __launch_bounds__(256) void scan_pass1(
    const float* __restrict__ a, const float* __restrict__ b,
    float* __restrict__ Aprod, float* __restrict__ U,
    int Bn, int T, int N, int L)
{
    const int CH = Bn * N;
    const int P  = T / L;
    int tid = blockIdx.x * blockDim.x + threadIdx.x;
    if (tid >= CH * P) return;
    const int ch = tid % CH;           // consecutive threads -> coalesced over n
    const int p  = tid / CH;
    const int bi = ch / N;
    const int n  = ch % N;
    size_t base = ((size_t)bi * T + (size_t)p * L) * N + n;
    float Ap = 1.0f, Uv = 0.0f;
    for (int t = 0; t < L; ++t) {
        float at = a[base + (size_t)t * N];
        float bt = b[base + (size_t)t * N];
        Ap *= at;
        Uv = at * Uv + (1.0f - at) * bt;
    }
    Aprod[(size_t)p * CH + ch] = Ap;
    U    [(size_t)p * CH + ch] = Uv;
}

__global__ __launch_bounds__(256) void scan_pass2(
    const float* __restrict__ Aprod, const float* __restrict__ U,
    float* __restrict__ Sinit, int CH, int P)
{
    int ch = blockIdx.x * blockDim.x + threadIdx.x;
    if (ch >= CH) return;
    float s = 0.0f;
    for (int p = 0; p < P; ++p) {
        Sinit[(size_t)p * CH + ch] = s;
        s = Aprod[(size_t)p * CH + ch] * s + U[(size_t)p * CH + ch];
    }
}

__global__ __launch_bounds__(256) void scan_pass3(
    const float* __restrict__ a, const float* __restrict__ b,
    const float* __restrict__ c, const float* __restrict__ Sinit,
    u16* __restrict__ cs, int Bn, int T, int N, int L)
{
    const int CH = Bn * N;
    const int P  = T / L;
    int tid = blockIdx.x * blockDim.x + threadIdx.x;
    if (tid >= CH * P) return;
    const int ch = tid % CH;
    const int p  = tid / CH;
    const int bi = ch / N;
    const int n  = ch % N;
    size_t base = ((size_t)bi * T + (size_t)p * L) * N + n;
    float s = Sinit[(size_t)p * CH + ch];
    for (int t = 0; t < L; ++t) {
        size_t idx = base + (size_t)t * N;
        float at = a[idx];
        float bt = b[idx];
        s = at * s + (1.0f - at) * bt;
        cs[idx] = f2bf(c[idx] * s);
    }
}

// ---------------------------------------------------------------------------
// Host launcher
// ---------------------------------------------------------------------------
extern "C" void kernel_launch(void* const* d_in, const int* in_sizes, int n_in,
                              void* d_out, int out_size, void* d_ws, size_t ws_size,
                              hipStream_t stream)
{
    (void)in_sizes; (void)n_in; (void)out_size; (void)ws_size;

    const float* x  = (const float*)d_in[0];
    const float* Wa = (const float*)d_in[1];
    const float* ba = (const float*)d_in[2];
    const float* Wb = (const float*)d_in[3];
    const float* bb = (const float*)d_in[4];
    const float* Wc = (const float*)d_in[5];
    const float* bc = (const float*)d_in[6];
    const float* Wd = (const float*)d_in[7];
    const float* bd = (const float*)d_in[8];
    const float* Wy = (const float*)d_in[9];
    const float* by = (const float*)d_in[10];

    constexpr int Bn = 8, T = 2048, D = 1024, N = 256;
    constexpr int M  = Bn * T;          // 16384
    constexpr int L  = 128;             // scan chunk length
    constexpr int P  = T / L;           // 16 chunks
    constexpr int CH = Bn * N;          // 2048 channels

    // Workspace carve-up (256B aligned)
    unsigned char* ws = (unsigned char*)d_ws;
    size_t off = 0;
    auto alloc = [&](size_t bytes) -> void* {
        void* p = (void*)(ws + off);
        off += bytes;
        off = (off + 255) & ~(size_t)255;
        return p;
    };
    u16*   xbf   = (u16*)  alloc((size_t)M * D * 2);    // x in bf16
    u16*   Wat   = (u16*)  alloc((size_t)N * D * 2);    // Wa^T bf16 (N x D)
    u16*   Wbt   = (u16*)  alloc((size_t)N * D * 2);
    u16*   Wct   = (u16*)  alloc((size_t)N * D * 2);
    u16*   Wdt   = (u16*)  alloc((size_t)D * D * 2);    // Wd^T bf16 (D x D)
    u16*   Wyt   = (u16*)  alloc((size_t)D * N * 2);    // Wy^T bf16 (D x N)
    float* aBuf  = (float*)alloc((size_t)M * N * 4);
    float* bBuf  = (float*)alloc((size_t)M * N * 4);
    float* cBuf  = (float*)alloc((size_t)M * N * 4);
    u16*   csBf  = (u16*)  alloc((size_t)M * N * 2);
    float* Aprod = (float*)alloc((size_t)CH * P * 4);
    float* Ubuf  = (float*)alloc((size_t)CH * P * 4);
    float* Sinit = (float*)alloc((size_t)CH * P * 4);

    float* y = (float*)d_out;           // also holds d-projection (M x D)

    // --- conversions ---
    convert_f32_bf16<<<1024, 256, 0, stream>>>(x, xbf, M * D);
    transpose_to_bf16<<<1024, 256, 0, stream>>>(Wa, Wat, D, N);
    transpose_to_bf16<<<1024, 256, 0, stream>>>(Wb, Wbt, D, N);
    transpose_to_bf16<<<1024, 256, 0, stream>>>(Wc, Wct, D, N);
    transpose_to_bf16<<<1024, 256, 0, stream>>>(Wd, Wdt, D, D);
    transpose_to_bf16<<<1024, 256, 0, stream>>>(Wy, Wyt, N, D);

    // --- gate / projection GEMMs (block tile 128x128) ---
    dim3 blk(256);
    dim3 gN(M / 128, N / 128);  // 128 x 2
    dim3 gD(M / 128, D / 128);  // 128 x 8
    gemm_bf16_wmma<<<gN, blk, 0, stream>>>(xbf, Wat, ba, nullptr, aBuf, M, N, D, 1); // sigmoid
    gemm_bf16_wmma<<<gN, blk, 0, stream>>>(xbf, Wbt, bb, nullptr, bBuf, M, N, D, 0);
    gemm_bf16_wmma<<<gN, blk, 0, stream>>>(xbf, Wct, bc, nullptr, cBuf, M, N, D, 2); // tanh
    gemm_bf16_wmma<<<gD, blk, 0, stream>>>(xbf, Wdt, bd, nullptr, y,    M, D, D, 0); // d -> out

    // --- chunked linear scan ---
    scan_pass1<<<(CH * P + 255) / 256, blk, 0, stream>>>(aBuf, bBuf, Aprod, Ubuf, Bn, T, N, L);
    scan_pass2<<<(CH + 255) / 256,      blk, 0, stream>>>(Aprod, Ubuf, Sinit, CH, P);
    scan_pass3<<<(CH * P + 255) / 256, blk, 0, stream>>>(aBuf, bBuf, cBuf, Sinit, csBf, Bn, T, N, L);

    // --- output projection: y = cs @ Wy + by + d (residual aliases out) ---
    gemm_bf16_wmma<<<gD, blk, 0, stream>>>(csBf, Wyt, by, y, y, M, D, N, 0);
}
